// _FusedMLPWrapper_87849261072828
// MI455X (gfx1250) — matrix operand order repaired and checked by reading, not compiled, not run on hardware
//
#include <hip/hip_runtime.h>

// ---------------------------------------------------------------------------
// Fused MLP + rank-32 LoRA for MI455X (gfx1250, wave32, WMMA 16x16x32 f16)
//   h   = gelu_tanh( x @ W1^T + (x @ ld1^T) @ lu1^T + b1 )
//   out =            h @ W2^T + (h @ ld2^T) @ lu2^T + b2
// Weight streams (268 MB total) are the roofline-critical traffic; the main
// GEMM double-buffers them through LDS with GLOBAL_LOAD_ASYNC_TO_LDS_B128
// (ASYNCcnt-tracked), overlapping the HBM stream with WMMA issue.
// ---------------------------------------------------------------------------

typedef __attribute__((ext_vector_type(16))) _Float16 v16h;
typedef __attribute__((ext_vector_type(8)))  _Float16 v8h;
typedef __attribute__((ext_vector_type(8)))  float    v8f;

#if defined(__AMDGCN__) && __has_builtin(__builtin_amdgcn_global_load_async_to_lds_b128)
#define USE_ASYNC 1
#else
#define USE_ASYNC 0
#endif

// Builtin signature (from the compiler's own diagnostic):
//   void __builtin_amdgcn_global_load_async_to_lds_b128(
//       int4 addrspace(1)* src, int4 addrspace(3)* dst, imm int offset, imm int cpol)
typedef int v4i_gcc __attribute__((__vector_size__(4 * sizeof(int))));
typedef __attribute__((address_space(1))) v4i_gcc* gb128_t;   // global int4*
typedef __attribute__((address_space(3))) v4i_gcc* sb128_t;   // LDS    int4*

__device__ __forceinline__ gb128_t as_global_b128(const void* p) {
  // generic and global share the same 64-bit representation
  return (gb128_t)(uintptr_t)p;
}
__device__ __forceinline__ sb128_t as_shared_b128(const void* p) {
  // generic->LDS: low 32 bits of the generic address are the LDS byte offset
  return (sb128_t)(unsigned int)(uintptr_t)p;
}

template<int N>
__device__ __forceinline__ void wait_async() {
#if defined(__AMDGCN__) && __has_builtin(__builtin_amdgcn_s_wait_asynccnt)
  __builtin_amdgcn_s_wait_asynccnt((short)N);
#else
  asm volatile("s_wait_asynccnt %0" :: "i"(N) : "memory");
#endif
}

union FragU { v16h v; v8h h[2]; };

// A fragment: 16(M) x 32(K) f16.
// ISA layout: lanes 0-15 hold row M=lane with K={0..7,16..23};
//             lanes 16-31 hold row M=lane-16 with K={8..15,24..31}.
__device__ __forceinline__ v16h load_fragA(const _Float16* __restrict__ base, int stride) {
  const int lane = threadIdx.x & 31;
  const int r    = lane & 15;
  const int kh   = (lane >> 4) << 3;           // 0 or 8
  const _Float16* p = base + r * stride + kh;
  FragU f;
  f.h[0] = *(const v8h*)(p);                   // K = kh .. kh+7
  f.h[1] = *(const v8h*)(p + 16);              // K = kh+16 .. kh+23
  return f.v;
}

// B fragment: 32(K) x 16(N), sourced from row-major [N,K] rows (W / lu / ld).
// ISA layout: lanes 0-15 hold col N=lane with K={0..7,16..23};
//             lanes 16-31 hold col N=lane-16 with K={8..15,24..31}.
__device__ __forceinline__ v16h load_fragB(const _Float16* __restrict__ base, int stride) {
  const int lane = threadIdx.x & 31;
  const int n    = lane & 15;
  const int k0   = (lane >> 4) << 3;           // 0 or 8
  const _Float16* p = base + n * stride + k0;
  FragU f;
  f.h[0] = *(const v8h*)(p);
  f.h[1] = *(const v8h*)(p + 16);
  return f.v;
}

__device__ __forceinline__ float gelu_tanh_f32(float x) {
  const float c0 = 0.7978845608028654f;        // sqrt(2/pi)
  const float c1 = 0.044715f;
  float u = c0 * (x + c1 * x * x * x);
  return 0.5f * x * (1.0f + tanhf(u));
}

#define BM  128
#define BN  128
#define BK  32
#define LDT 40   // padded LDS row stride in halves (80 B) -> conflict-free frag reads
// async-to-LDS ops issued per thread per (A+B) stage: (BM*4 + BN*4)/256
#define STAGE_OPS 4

// Synchronous staging fallback (global -> VGPR -> ds_store), with prefetch hint.
template<int ROWS>
__device__ __forceinline__ void stage_tile_sync(const _Float16* __restrict__ g, int ldg,
                                                _Float16* __restrict__ lds, bool pf) {
  for (int c = threadIdx.x; c < ROWS * 4; c += 256) {
    const int r = c >> 2;
    const int k = (c & 3) << 3;
    *(v8h*)(lds + r * LDT + k) = *(const v8h*)(g + (size_t)r * ldg + k);
    if (pf) __builtin_prefetch((const void*)(g + (size_t)r * ldg + k + BK), 0, 1);
  }
}

#if USE_ASYNC
// Asynchronous staging: GLOBAL_LOAD_ASYNC_TO_LDS_B128, tracked by ASYNCcnt.
template<int ROWS>
__device__ __forceinline__ void stage_tile_async(const _Float16* __restrict__ g, int ldg,
                                                 _Float16* __restrict__ lds) {
  for (int c = threadIdx.x; c < ROWS * 4; c += 256) {
    const int r = c >> 2;
    const int k = (c & 3) << 3;
    __builtin_amdgcn_global_load_async_to_lds_b128(
        as_global_b128(g + (size_t)r * ldg + k),
        as_shared_b128(lds + r * LDT + k), 0, 0);
  }
}
#endif

// ---------------------------------------------------------------------------
// Main fused linear: Y = act( A[M,K] @ W[N,K]^T + T[M,32] @ LU[N,32]^T + bias )
// Block: 256 threads (8 waves, 4x2). Block tile 128x128, wave tile 32x64:
// each B fragment is reused by two A fragments (8 WMMA per 12 ds_loads).
// Double-buffered LDS pipeline: stage tile i+1 async, wait own ASYNCcnt so
// tile i's (earlier, in-order) group is complete, barrier, WMMA on tile i.
// ---------------------------------------------------------------------------
template<bool GELU>
__global__ __launch_bounds__(256, 2)
void fused_linear_kernel(const _Float16* __restrict__ A,
                         const _Float16* __restrict__ W,
                         const _Float16* __restrict__ bias,
                         const _Float16* __restrict__ T,    // lora-down result [M,32]
                         const _Float16* __restrict__ LU,   // lora-up          [N,32]
                         _Float16* __restrict__ Y,
                         int M, int N, int K) {
  __shared__ __align__(16) _Float16 sA[2][BM * LDT];
  __shared__ __align__(16) _Float16 sB[2][BN * LDT];

  const int bm = blockIdx.y * BM;
  const int bn = blockIdx.x * BN;
  const int waveid = threadIdx.x >> 5;   // wave32
  const int wm = waveid >> 1;            // 0..3 -> 32-row slice
  const int wn = waveid & 1;             // 0..1 -> 64-col slice

  const _Float16* gA = A + (size_t)bm * K;
  const _Float16* gW = W + (size_t)bn * K;

  v8f acc[2][4] = {};

#if USE_ASYNC
  const int NT = K / BK;
  stage_tile_async<BM>(gA, K, sA[0]);
  stage_tile_async<BN>(gW, K, sB[0]);
  for (int it = 0; it < NT; ++it) {
    const int cur = it & 1;
    if (it + 1 < NT) {
      stage_tile_async<BM>(gA + (size_t)(it + 1) * BK, K, sA[cur ^ 1]);
      stage_tile_async<BN>(gW + (size_t)(it + 1) * BK, K, sB[cur ^ 1]);
      wait_async<STAGE_OPS>();   // group for tile 'it' (issued earlier) is done
    } else {
      wait_async<0>();
    }
    __syncthreads();             // every wave has waited on its own copies

    const v16h a0 = load_fragA(sA[cur] + (wm * 32     ) * LDT, LDT);
    const v16h a1 = load_fragA(sA[cur] + (wm * 32 + 16) * LDT, LDT);
#pragma unroll
    for (int s = 0; s < 4; ++s) {
      const v16h b = load_fragB(sB[cur] + (wn * 64 + s * 16) * LDT, LDT);
      acc[0][s] = __builtin_amdgcn_wmma_f32_16x16x32_f16(
          false, a0, false, b, (short)0, acc[0][s], false, false);
      acc[1][s] = __builtin_amdgcn_wmma_f32_16x16x32_f16(
          false, a1, false, b, (short)0, acc[1][s], false, false);
    }
    __syncthreads();             // all waves done reading before buffer reuse
  }
#else
  for (int kt = 0; kt < K; kt += BK) {
    const bool pf = (kt + BK) < K;
    stage_tile_sync<BM>(gA + kt, K, sA[0], pf);
    stage_tile_sync<BN>(gW + kt, K, sB[0], pf);
    __syncthreads();

    const v16h a0 = load_fragA(sA[0] + (wm * 32     ) * LDT, LDT);
    const v16h a1 = load_fragA(sA[0] + (wm * 32 + 16) * LDT, LDT);
#pragma unroll
    for (int s = 0; s < 4; ++s) {
      const v16h b = load_fragB(sB[0] + (wn * 64 + s * 16) * LDT, LDT);
      acc[0][s] = __builtin_amdgcn_wmma_f32_16x16x32_f16(
          false, a0, false, b, (short)0, acc[0][s], false, false);
      acc[1][s] = __builtin_amdgcn_wmma_f32_16x16x32_f16(
          false, a1, false, b, (short)0, acc[1][s], false, false);
    }
    __syncthreads();
  }
#endif

  // LoRA rank-32 up-projection: exactly one extra WMMA per 16x16 tile.
  const v16h at0 = load_fragA(T + (size_t)(bm + wm * 32     ) * 32, 32);
  const v16h at1 = load_fragA(T + (size_t)(bm + wm * 32 + 16) * 32, 32);
#pragma unroll
  for (int s = 0; s < 4; ++s) {
    const v16h bl = load_fragB(LU + (size_t)(bn + wn * 64 + s * 16) * 32, 32);
    acc[0][s] = __builtin_amdgcn_wmma_f32_16x16x32_f16(
        false, at0, false, bl, (short)0, acc[0][s], false, false);
    acc[1][s] = __builtin_amdgcn_wmma_f32_16x16x32_f16(
        false, at1, false, bl, (short)0, acc[1][s], false, false);
  }

  // Epilogue: bias (+ GELU), f32 -> f16 store.
  // C/D layout: lanes 0-15 -> rows 0..7 (vgpr r), col=lane; lanes 16-31 -> rows 8..15.
  const int lane  = threadIdx.x & 31;
  const int col   = lane & 15;
  const int rbase = (lane >> 4) << 3;
#pragma unroll
  for (int s = 0; s < 4; ++s) {
    const int gn = bn + wn * 64 + s * 16 + col;
    const float bv = (float)bias[gn];
#pragma unroll
    for (int g = 0; g < 2; ++g) {
      const int grow = bm + wm * 32 + g * 16 + rbase;
#pragma unroll
      for (int r = 0; r < 8; ++r) {
        float v = acc[g][s][r] + bv;
        if (GELU) v = gelu_tanh_f32(v);
        Y[(size_t)(grow + r) * N + gn] = (_Float16)v;
      }
    }
  }
}

// ---------------------------------------------------------------------------
// LoRA down-projection: T[512,32] = S[512,K] @ D[32,K]^T.
// 64 waves total: wave -> (m-tile 0..31, rank-half 0..1); 1 WMMA per K-step.
// ---------------------------------------------------------------------------
__global__ __launch_bounds__(256, 4)
void lora_down_kernel(const _Float16* __restrict__ S,
                      const _Float16* __restrict__ D,
                      _Float16* __restrict__ T,
                      int K) {
  const int w  = blockIdx.x * 8 + (threadIdx.x >> 5);
  const int mt = w >> 1;
  const int nt = w & 1;
  const _Float16* srow = S + (size_t)mt * 16 * K;
  const _Float16* drow = D + (size_t)nt * 16 * K;

  v8f acc = {};
  for (int k = 0; k < K; k += 32) {
    const v16h a = load_fragA(srow + k, K);
    const v16h b = load_fragB(drow + k, K);
    acc = __builtin_amdgcn_wmma_f32_16x16x32_f16(
        false, a, false, b, (short)0, acc, false, false);
  }

  const int lane  = threadIdx.x & 31;
  const int col   = lane & 15;
  const int rbase = (lane >> 4) << 3;
#pragma unroll
  for (int r = 0; r < 8; ++r)
    T[(size_t)(mt * 16 + rbase + r) * 32 + nt * 16 + col] = (_Float16)acc[r];
}

// ---------------------------------------------------------------------------
extern "C" void kernel_launch(void* const* d_in, const int* in_sizes, int n_in,
                              void* d_out, int out_size, void* d_ws, size_t ws_size,
                              hipStream_t stream) {
  (void)in_sizes; (void)n_in; (void)out_size; (void)ws_size;

  const int TOK = 512, IN = 4096, HID = 16384;

  const _Float16* x   = (const _Float16*)d_in[0];
  const _Float16* W1  = (const _Float16*)d_in[1];
  const _Float16* b1  = (const _Float16*)d_in[2];
  const _Float16* ld1 = (const _Float16*)d_in[3];
  const _Float16* lu1 = (const _Float16*)d_in[4];
  const _Float16* W2  = (const _Float16*)d_in[5];
  const _Float16* b2  = (const _Float16*)d_in[6];
  const _Float16* ld2 = (const _Float16*)d_in[7];
  const _Float16* lu2 = (const _Float16*)d_in[8];
  _Float16* out = (_Float16*)d_out;

  char* ws = (char*)d_ws;
  _Float16* t1 = (_Float16*)(ws);             // [512,32]  32 KB
  _Float16* t2 = (_Float16*)(ws + 32768);     // [512,32]  32 KB
  _Float16* h  = (_Float16*)(ws + 65536);     // [512,16384] 16 MB

  // t1 = x @ ld1^T
  lora_down_kernel<<<8, 256, 0, stream>>>(x, ld1, t1, IN);

  // h = gelu(x @ W1^T + t1 @ lu1^T + b1)
  fused_linear_kernel<true><<<dim3(HID / BN, TOK / BM), 256, 0, stream>>>(
      x, W1, b1, t1, lu1, h, TOK, HID, IN);

  // t2 = h @ ld2^T
  lora_down_kernel<<<8, 256, 0, stream>>>(h, ld2, t2, HID);

  // out = h @ W2^T + t2 @ lu2^T + b2
  fused_linear_kernel<false><<<dim3(IN / BN, TOK / BM), 256, 0, stream>>>(
      h, W2, b2, t2, lu2, out, TOK, IN, HID);
}
